// DeCANAttention_43533788512525
// MI455X (gfx1250) — matrix operand order repaired and checked by reading, not compiled, not run on hardware
//
#include <hip/hip_runtime.h>
#include <hip/hip_bf16.h>

// ---------------------------------------------------------------------------
// DeCAN attention for MI455X (gfx1250), wave32, WMMA f16->f32,
// LDS-staged GEMMs with gfx1250 async global->LDS copies (double buffered).
//   B=2, L=1024, H=2048, HD=64, NK=4, NQ=32, NPREV=28
// ---------------------------------------------------------------------------

typedef _Float16 f16;
typedef __attribute__((ext_vector_type(16))) _Float16 v16h;
typedef __attribute__((ext_vector_type(8)))  _Float16 v8h;
typedef __attribute__((ext_vector_type(8)))  float    v8f;
typedef __attribute__((ext_vector_type(4)))  int      v4i;

#define BB   2
#define LL   1024
#define HH   2048
#define HD   64
#define NQ   32
#define NK   4
#define NPREV 28

// D = A(16x32 f16) * B(32x16 f16) + C(16x16 f32)
__device__ __forceinline__ v8f wmma_f16(v16h a, v16h b, v8f c) {
  return __builtin_amdgcn_wmma_f32_16x16x32_f16(
      /*neg_a=*/false, a, /*neg_b=*/false, b,
      /*c_mod=*/(short)0, c, /*reuse_a=*/false, /*reuse_b=*/false);
}

// gfx1250 async copy: 16B global -> LDS, tracked by ASYNCcnt.
// Builtin signature (from hipcc diagnostic): arg0 = v4i addrspace(1)* (src),
// arg1 = v4i addrspace(3)* (LDS dst), then i32 offset, i32 cpol.
__device__ __forceinline__ void async_copy16(f16* lds_dst, const f16* g_src) {
#if defined(__gfx1250__) && \
    __has_builtin(__builtin_amdgcn_global_load_async_to_lds_b128)
  __builtin_amdgcn_global_load_async_to_lds_b128(
      (__attribute__((address_space(1))) v4i*)(f16*)g_src,
      (__attribute__((address_space(3))) v4i*)lds_dst,
      /*offset=*/0, /*cpol=*/0);
#else
  *(v8h*)lds_dst = *(const v8h*)g_src;   // synchronous fallback
#endif
}

__device__ __forceinline__ void wait_async0() {
#if defined(__gfx1250__) && __has_builtin(__builtin_amdgcn_s_wait_asynccnt)
  __builtin_amdgcn_s_wait_asynccnt(0);
#elif defined(__gfx1250__) && \
    __has_builtin(__builtin_amdgcn_global_load_async_to_lds_b128)
  asm volatile("s_wait_asynccnt 0x0" ::: "memory");
#endif
}

// A-fragment: row-major X[.][ld], tile rows [row0,row0+16), K cols [k0,k0+32)
// lane m=lane&15 holds row m; element e holds K = half*16 + e
__device__ __forceinline__ v16h load_a_frag(const f16* X, int row0, int k0,
                                            int ld, int lane) {
  int m = lane & 15, h = lane >> 4;
  const f16* p = X + (size_t)(row0 + m) * ld + (k0 + h * 16);
  v8h lo = *(const v8h*)p;          // K = k0 + h*16 .. +7
  v8h hi = *(const v8h*)(p + 8);    // K = k0 + h*16 + 8 .. +15
  return __builtin_shufflevector(lo, hi, 0, 1, 2, 3, 4, 5, 6, 7,
                                 8, 9, 10, 11, 12, 13, 14, 15);
}

// B-fragment where logical B[k][n] = W[n][k] (W row-major, leading dim ld).
// lane n=lane&15 holds column n; element e holds K = half*16 + e.
__device__ __forceinline__ v16h load_b_rowk(const f16* W, int col0, int k0,
                                            int ld, int lane) {
  int n = lane & 15, h = lane >> 4;
  return *(const v16h*)(W + (size_t)(col0 + n) * ld + (k0 + h * 16));
}

// ---------------------------------------------------------------------------
// 1) f32 -> f16 conversion (grid-stride)
// ---------------------------------------------------------------------------
__global__ void cvt_f32_to_f16(const float* __restrict__ src,
                               f16* __restrict__ dst, int n) {
  for (int i = blockIdx.x * blockDim.x + threadIdx.x; i < n;
       i += gridDim.x * blockDim.x)
    dst[i] = (f16)src[i];
}

// ---------------------------------------------------------------------------
// 2) GEMM: C[M][N] = A[M][K] * W[N][K]^T   (f16 in, f32 out)
//    Block = 4 waves = 64x64 C tile. A/B K-slices staged in LDS via async
//    copies, double buffered; B tile shared by all 4 waves.
// ---------------------------------------------------------------------------
#define KSTEP 32
__global__ __launch_bounds__(128) void gemm_f16_lds(const f16* __restrict__ A,
                                                    const f16* __restrict__ W,
                                                    float* __restrict__ C,
                                                    int M, int N, int K) {
  __shared__ __attribute__((aligned(64))) f16 Abuf[2][64 * KSTEP];
  __shared__ __attribute__((aligned(64))) f16 Bbuf[2][64 * KSTEP];

  const int t = threadIdx.x;
  const int lane = t & 31, wave = t >> 5;
  const int row0 = blockIdx.y * 64;
  const int col0 = blockIdx.x * 64;

  // Stage one 64x32 A slab + 64x32 B slab (8KB) : 128 threads x 2 x 16B each.
  auto stage = [&](int buf, int k0) {
#pragma unroll
    for (int i = 0; i < 2; ++i) {
      int c = t + i * 128;           // chunk 0..255
      int row = c >> 2;              // 0..63
      int seg = (c & 3) * 8;         // f16 offset within 32-wide K slice
      async_copy16(&Abuf[buf][row * KSTEP + seg],
                   A + (size_t)(row0 + row) * K + k0 + seg);
      async_copy16(&Bbuf[buf][row * KSTEP + seg],
                   W + (size_t)(col0 + row) * K + k0 + seg);
    }
  };

  v8f acc[4] = {v8f{}, v8f{}, v8f{}, v8f{}};
  const int nk = K / KSTEP;
  stage(0, 0);
  for (int kt = 0; kt < nk; ++kt) {
    wait_async0();
    __syncthreads();                 // slab kt visible to all waves
    const int buf = kt & 1;
    if (kt + 1 < nk) stage(buf ^ 1, (kt + 1) * KSTEP);  // overwrites slab kt-1
    v16h a = load_a_frag(&Abuf[buf][wave * 16 * KSTEP], 0, 0, KSTEP, lane);
#pragma unroll
    for (int j = 0; j < 4; ++j) {
      v16h b = load_b_rowk(&Bbuf[buf][0], 16 * j, 0, KSTEP, lane);
      acc[j] = wmma_f16(a, b, acc[j]);
    }
  }

  const int n = lane & 15, h = lane >> 4;
  const int wrow = row0 + wave * 16;
#pragma unroll
  for (int j = 0; j < 4; ++j)
#pragma unroll
    for (int r = 0; r < 8; ++r)
      C[(size_t)(wrow + h * 8 + r) * N + (col0 + 16 * j + n)] = acc[j][r];
}

// ---------------------------------------------------------------------------
// 3) RoPE + cross-layer KV concat + pack to f16
//    idx over [B][NQ][L][HD]; q16/k16 -> [b,h,l,d], vT16 -> [b,h,d,l]
// ---------------------------------------------------------------------------
__global__ void rope_pack(const float* __restrict__ qp,     // [B*L][NQ*HD]
                          const float* __restrict__ kp,     // [B*L][NK*HD]
                          const float* __restrict__ vp,     // [B*L][NK*HD]
                          const float* __restrict__ prev_k, // [B][28][L][HD]
                          const float* __restrict__ prev_v,
                          const float* __restrict__ cosb,   // [L][HD]
                          const float* __restrict__ sinb,
                          f16* __restrict__ q16, f16* __restrict__ k16,
                          f16* __restrict__ vt16) {
  int idx = blockIdx.x * blockDim.x + threadIdx.x; // B*NQ*L*HD = 4194304
  int d  = idx & (HD - 1);
  int l  = (idx >> 6) & (LL - 1);
  int hh = (idx >> 16) & (NQ - 1);
  int b  = idx >> 21;

  float cv = cosb[l * HD + d];
  float sv = sinb[l * HD + d];
  int   dp = (d < HD / 2) ? d + HD / 2 : d - HD / 2;
  float sg = (d < HD / 2) ? -1.0f : 1.0f;

  // Q
  size_t qb = (size_t)(b * LL + l) * (NQ * HD) + hh * HD;
  float qv = qp[qb + d];
  float qr = sg * qp[qb + dp];
  q16[idx] = (f16)(qv * cv + qr * sv);

  // K/V = concat(prev, new)[h]
  float kv, kr, vv;
  if (hh < NPREV) {
    size_t pb = (size_t)((b * NPREV + hh) * LL + l) * HD;
    kv = prev_k[pb + d];
    kr = sg * prev_k[pb + dp];
    vv = prev_v[pb + d];
  } else {
    size_t nb = (size_t)(b * LL + l) * (NK * HD) + (hh - NPREV) * HD;
    kv = kp[nb + d];
    kr = sg * kp[nb + dp];
    vv = vp[nb + d];
  }
  k16[idx] = (f16)(kv * cv + kr * sv);
  vt16[(size_t)((b * NQ + hh) * HD + d) * LL + l] = (f16)vv;
}

// ---------------------------------------------------------------------------
// 4) Flash attention: one wave per 16-row q tile per (b,h).
//    Scores + PV via WMMA; online softmax with shfl row-reductions;
//    P converted C-layout -> A-layout through LDS.
// ---------------------------------------------------------------------------
__global__ __launch_bounds__(128) void attn_kernel(const f16* __restrict__ Q,
                                                   const f16* __restrict__ Kc,
                                                   const f16* __restrict__ Vt,
                                                   f16* __restrict__ Oa) {
  __shared__ __attribute__((aligned(32))) f16 lds_p[4][16 * 32];

  int lane = threadIdx.x & 31;
  int wave = threadIdx.x >> 5;
  int tile = blockIdx.x * 4 + wave;       // B*NQ*(L/16) tiles
  int qt = tile & (LL / 16 - 1);          // 64 q tiles
  int hh = (tile >> 6) & (NQ - 1);
  int b  = tile >> 11;
  int qb = qt * 16;
  int half = lane >> 4, n = lane & 15;

  const f16* q  = Q  + (size_t)(b * NQ + hh) * LL * HD;
  const f16* k  = Kc + (size_t)(b * NQ + hh) * LL * HD;
  const f16* vt = Vt + (size_t)(b * NQ + hh) * HD * LL;
  f16* myp = lds_p[wave];

  v16h aq0 = load_a_frag(q, qb, 0, HD, lane);
  v16h aq1 = load_a_frag(q, qb, 32, HD, lane);

  v8f o[4] = {v8f{}, v8f{}, v8f{}, v8f{}};
  float mrun[8], lrun[8];
#pragma unroll
  for (int r = 0; r < 8; ++r) { mrun[r] = -1e30f; lrun[r] = 0.0f; }

  int nblocks = (qb + 47) >> 5;           // causal: cover kv cols <= qb+15
  for (int kb = 0; kb < nblocks * 32; kb += 32) {
    // S tile 16 x 32 = two 16x16 frags, K-dim = HD in two chunks of 32
    v8f s0 = v8f{}, s1 = v8f{};
    {
      v16h b0 = load_b_rowk(k, kb,      0,  HD, lane);
      v16h b1 = load_b_rowk(k, kb,      32, HD, lane);
      s0 = wmma_f16(aq0, b0, s0);
      s0 = wmma_f16(aq1, b1, s0);
      v16h b2 = load_b_rowk(k, kb + 16, 0,  HD, lane);
      v16h b3 = load_b_rowk(k, kb + 16, 32, HD, lane);
      s1 = wmma_f16(aq0, b2, s1);
      s1 = wmma_f16(aq1, b3, s1);
    }
#pragma unroll
    for (int r = 0; r < 8; ++r) {
      int qi = qb + half * 8 + r;
      float x0 = s0[r] * 0.125f;                 // 1/sqrt(64)
      float x1 = s1[r] * 0.125f;
      if (kb + n > qi)       x0 = -1e30f;        // causal mask
      if (kb + 16 + n > qi)  x1 = -1e30f;
      float vmax = fmaxf(x0, x1);
#pragma unroll
      for (int mm = 1; mm < 16; mm <<= 1)
        vmax = fmaxf(vmax, __shfl_xor(vmax, mm));
      float mnew  = fmaxf(mrun[r], vmax);
      float alpha = __expf(mrun[r] - mnew);
      float p0 = __expf(x0 - mnew);
      float p1 = __expf(x1 - mnew);
      float rs = p0 + p1;
#pragma unroll
      for (int mm = 1; mm < 16; mm <<= 1)
        rs += __shfl_xor(rs, mm);
      lrun[r] = lrun[r] * alpha + rs;
      mrun[r] = mnew;
#pragma unroll
      for (int f = 0; f < 4; ++f) o[f][r] *= alpha;
      // stash P in LDS (row = half*8+r, cols kb..kb+31 within tile)
      myp[(half * 8 + r) * 32 + n]      = (f16)p0;
      myp[(half * 8 + r) * 32 + 16 + n] = (f16)p1;
    }
    // reload P as A-fragment (16x32) from LDS
    v8h plo = *(const v8h*)(myp + n * 32 + half * 16);
    v8h phi = *(const v8h*)(myp + n * 32 + half * 16 + 8);
    v16h pa = __builtin_shufflevector(plo, phi, 0, 1, 2, 3, 4, 5, 6, 7,
                                      8, 9, 10, 11, 12, 13, 14, 15);
    // O += P * V, V supplied transposed: B[kv][d] = Vt[d][kv]
#pragma unroll
    for (int f = 0; f < 4; ++f) {
      v16h bv = *(const v16h*)(vt + (size_t)(16 * f + n) * LL +
                               (kb + half * 16));
      o[f] = wmma_f16(pa, bv, o[f]);
    }
  }

  // epilogue: normalize by l and scatter to [b][qi][hh*64 + d] (f16)
#pragma unroll
  for (int f = 0; f < 4; ++f)
#pragma unroll
    for (int r = 0; r < 8; ++r) {
      int qi = qb + half * 8 + r;
      float val = o[f][r] / lrun[r];
      Oa[(size_t)(b * LL + qi) * (NQ * HD) + hh * HD + 16 * f + n] = (f16)val;
    }
}

// ---------------------------------------------------------------------------
// Host-side launch
// ---------------------------------------------------------------------------
extern "C" void kernel_launch(void* const* d_in, const int* in_sizes, int n_in,
                              void* d_out, int out_size, void* d_ws,
                              size_t ws_size, hipStream_t stream) {
  (void)in_sizes; (void)n_in; (void)out_size; (void)ws_size;

  const float* hs     = (const float*)d_in[0];
  const float* prev_k = (const float*)d_in[1];
  const float* prev_v = (const float*)d_in[2];
  const float* Wq     = (const float*)d_in[3];
  const float* Wk     = (const float*)d_in[4];
  const float* Wv     = (const float*)d_in[5];
  const float* Wo     = (const float*)d_in[6];
  const float* cosb   = (const float*)d_in[7];
  const float* sinb   = (const float*)d_in[8];
  float* out = (float*)d_out;

  const int ML  = BB * LL;           // 2048 rows
  const int nHS = ML * HH;           // 4194304
  const int nWq = NQ * HD * HH;      // 4194304
  const int nWk = NK * HD * HH;      // 524288
  const int nQKV = BB * NQ * LL * HD;// 4194304

  // workspace partition (all offsets 256B-aligned)
  char* ws = (char*)d_ws;
  size_t off = 0;
  auto take = [&](size_t bytes) {
    char* p = ws + off;
    off += (bytes + 255) & ~(size_t)255;
    return p;
  };
  f16*   hs16   = (f16*)take((size_t)nHS * 2);
  f16*   wq16   = (f16*)take((size_t)nWq * 2);
  f16*   wk16   = (f16*)take((size_t)nWk * 2);
  f16*   wv16   = (f16*)take((size_t)nWk * 2);
  f16*   wo16   = (f16*)take((size_t)nWq * 2);
  float* qp     = (float*)take((size_t)ML * NQ * HD * 4);
  float* kp     = (float*)take((size_t)ML * NK * HD * 4);
  float* vp     = (float*)take((size_t)ML * NK * HD * 4);
  f16*   q16    = (f16*)take((size_t)nQKV * 2);
  f16*   k16    = (f16*)take((size_t)nQKV * 2);
  f16*   vt16   = (f16*)take((size_t)nQKV * 2);
  f16*   attn16 = (f16*)take((size_t)ML * NQ * HD * 2);

  // 1) convert to f16
  cvt_f32_to_f16<<<(nHS + 255) / 256, 256, 0, stream>>>(hs, hs16, nHS);
  cvt_f32_to_f16<<<(nWq + 255) / 256, 256, 0, stream>>>(Wq, wq16, nWq);
  cvt_f32_to_f16<<<(nWk + 255) / 256, 256, 0, stream>>>(Wk, wk16, nWk);
  cvt_f32_to_f16<<<(nWk + 255) / 256, 256, 0, stream>>>(Wv, wv16, nWk);
  cvt_f32_to_f16<<<(nWq + 255) / 256, 256, 0, stream>>>(Wo, wo16, nWq);

  // 2) QKV projections (WMMA, async-LDS staged)
  gemm_f16_lds<<<dim3(NQ * HD / 64, ML / 64), 128, 0, stream>>>(
      hs16, wq16, qp, ML, NQ * HD, HH);
  gemm_f16_lds<<<dim3(NK * HD / 64, ML / 64), 128, 0, stream>>>(
      hs16, wk16, kp, ML, NK * HD, HH);
  gemm_f16_lds<<<dim3(NK * HD / 64, ML / 64), 128, 0, stream>>>(
      hs16, wv16, vp, ML, NK * HD, HH);

  // 3) RoPE + concat + pack
  rope_pack<<<nQKV / 256, 256, 0, stream>>>(qp, kp, vp, prev_k, prev_v,
                                            cosb, sinb, q16, k16, vt16);

  // 4) flash attention (one wave per 16-row q tile; 4 waves / block)
  int tiles = BB * NQ * (LL / 16);   // 4096
  attn_kernel<<<tiles / 4, 128, 0, stream>>>(q16, k16, vt16, attn16);

  // 5) output projection straight into d_out (f32)
  gemm_f16_lds<<<dim3(HH / 64, ML / 64), 128, 0, stream>>>(
      attn16, wo16, out, ML, HH, NQ * HD);
}
